// InteractionLayer_76759655514801
// MI455X (gfx1250) — matrix operand order repaired and checked by reading, not compiled, not run on hardware
//
#include <hip/hip_runtime.h>
#include <math.h>

// ---------------------------------------------------------------------------
// MI455X / gfx1250 implementation of the interaction layer.
// All dense GEMMs use v_wmma_f32_16x16x32_f16 (f16 inputs, f32 accumulate).
// Segment softmax / segment sums use global float atomics.
// Input pointer order = setup_inputs() dict order with params flattened in
// insertion order (env_W[3], env_b[3], Wq, Wk, scl_W[3], scl_b[3], ln_g,
// ln_b, eq_W[3], eq_b[3], Ws, Wv).
// ---------------------------------------------------------------------------

typedef __attribute__((ext_vector_type(16))) _Float16 v16h;
typedef __attribute__((ext_vector_type(8)))  _Float16 v8h;
typedef __attribute__((ext_vector_type(8)))  float    v8f;

#define NNODES 10000
#define NEDGES 320000

// ---- WMMA fragment loaders (ISA 7.12.2 layouts) ---------------------------
// A (16x32 f16): lane L holds row m=L&15; halves 0..7 -> k = lo+h,
// halves 8..15 -> k = 16+lo+(h-8), lo = (L>=16)?8:0.  => two 16B loads.
__device__ __forceinline__ v16h load_a_frag(const _Float16* A, int lda,
                                            int row0, int kb, int lane) {
  const int m  = lane & 15;
  const int lo = (lane & 16) ? 8 : 0;
  const _Float16* p = A + (size_t)(row0 + m) * lda + kb + lo;
  v8h alo = *(const v8h*)(p);
  v8h ahi = *(const v8h*)(p + 16);
  v16h a;
#pragma unroll
  for (int i = 0; i < 8; ++i) { a[i] = alo[i]; a[i + 8] = ahi[i]; }
  return a;
}

// B (32x16 f16): lane L holds column n=L&15; halves 0..7 -> k = lo+h,
// halves 8..15 -> k = 16+lo+(h-8), lo = (L>=16)?8:0 (strided gather).
__device__ __forceinline__ v16h load_b_frag(const _Float16* B, int ldb,
                                            int kb, int n0, int lane) {
  const int n  = lane & 15;
  const int lo = (lane & 16) ? 8 : 0;
  const _Float16* p = B + (size_t)(kb + lo) * ldb + n0 + n;
  v16h b;
#pragma unroll
  for (int i = 0; i < 8; ++i) {
    b[i] = p[(size_t)i * ldb];
    b[i + 8] = p[(size_t)(i + 16) * ldb];
  }
  return b;
}

__device__ __forceinline__ float wsum16(float v) {
#pragma unroll
  for (int o = 1; o < 16; o <<= 1) v += __shfl_xor(v, o, 16);
  return v;
}
__device__ __forceinline__ float wsum32(float v) {
#pragma unroll
  for (int o = 1; o < 32; o <<= 1) v += __shfl_xor(v, o, 32);
  return v;
}

// Order-preserving float <-> uint key for atomicMax-based segment max.
__device__ __forceinline__ unsigned fkey(float f) {
  unsigned u = __float_as_uint(f);
  return (u & 0x80000000u) ? ~u : (u | 0x80000000u);
}
__device__ __forceinline__ float funkey(unsigned k) {
  unsigned u = (k & 0x80000000u) ? (k & 0x7fffffffu) : ~k;
  return __uint_as_float(u);
}
#define NEG_INF_KEY 0x007FFFFFu  // fkey(-inf)

// ---------------------------------------------------------------------------
// Generic WMMA GEMM:  out[E x N] = act( [A1|A2][E x (K1+K2)] @ B + bias )
// One wave computes one 16x16 tile.  4 waves / block cover 64 output cols.
// OUTMODE: 0 = f16 store, 1 = f32 store, 2 = atomicAdd(c_new*val) at
//          row active[row] (used for upd.at[active_edges].add()).
// ACT: 0 = none, 1 = SiLU.
// ---------------------------------------------------------------------------
template <int ACT, int OUTMODE>
__global__ void gemm_wmma(const _Float16* __restrict__ A1, int K1,
                          const _Float16* __restrict__ A2, int K2,
                          const _Float16* __restrict__ B,
                          const float* __restrict__ bias, int N,
                          _Float16* __restrict__ out16,
                          float* __restrict__ out32,
                          const int* __restrict__ active,
                          const float* __restrict__ ucoef) {
  const int lane = threadIdx.x & 31;
  const int wave = threadIdx.x >> 5;
  const int e0   = blockIdx.x * 16;
  const int n0   = (blockIdx.y * (blockDim.x >> 5) + wave) * 16;
  if (n0 >= N) return;

  v8f c = {};
  for (int kb = 0; kb < K1; kb += 32) {
    v16h a = load_a_frag(A1, K1, e0, kb, lane);
    v16h b = load_b_frag(B, N, kb, n0, lane);
    c = __builtin_amdgcn_wmma_f32_16x16x32_f16(false, a, false, b, (short)0, c,
                                               false, false);
  }
  if (A2) {
    const _Float16* B2 = B + (size_t)K1 * N;
    for (int kb = 0; kb < K2; kb += 32) {
      v16h a = load_a_frag(A2, K2, e0, kb, lane);
      v16h b = load_b_frag(B2, N, kb, n0, lane);
      c = __builtin_amdgcn_wmma_f32_16x16x32_f16(false, a, false, b, (short)0,
                                                 c, false, false);
    }
  }

  const int   n  = lane & 15;
  const int   rb = (lane & 16) ? 8 : 0;
  const float bb = bias ? bias[n0 + n] : 0.0f;
  float cn = 0.0f;
  if (OUTMODE == 2) {
    float u = ucoef[0];
    cn = u * rsqrtf(u * u + 1.0f);  // c_new
  }
#pragma unroll
  for (int r = 0; r < 8; ++r) {
    float v = c[r] + bb;
    if (ACT == 1) v = v / (1.0f + __expf(-v));  // SiLU
    const int row = e0 + rb + r;
    if (OUTMODE == 0)
      out16[(size_t)row * N + n0 + n] = (_Float16)v;
    else if (OUTMODE == 1)
      out32[(size_t)row * N + n0 + n] = v;
    else
      atomicAdd(&out32[(size_t)active[row] * N + n0 + n], cn * v);
  }
}

// ---------------------------------------------------------------------------
// Fused attention scores:  W[e,m] = 5 * sum_d Q[e,m,d]*K[e,m,d]
// K-tile (16 edges x 16 cols of sl@Wk) built in registers via 4 WMMAs, then
// multiplied by gathered Qn[c] (already in the C-matrix lane layout) and
// reduced across the 16 lanes of each half-wave.  Also segment-max via
// ordered-key atomicMax.
// ---------------------------------------------------------------------------
__global__ void score_wmma(const _Float16* __restrict__ sl16,
                           const _Float16* __restrict__ Wk16,
                           const float* __restrict__ Qn,
                           const int* __restrict__ ecen,
                           float* __restrict__ Wsc,
                           unsigned* __restrict__ nmax) {
  __shared__ int sc[16];
  const int lane = threadIdx.x & 31;
  const int wave = threadIdx.x >> 5;  // 0..3
  const int e0   = blockIdx.x * 16;
  if (threadIdx.x < 16) sc[threadIdx.x] = ecen[e0 + threadIdx.x];
  __syncthreads();

  // preload all four A fragments of the sl tile (K = 128)
  v16h a0 = load_a_frag(sl16, 128, e0, 0, lane);
  v16h a1 = load_a_frag(sl16, 128, e0, 32, lane);
  v16h a2 = load_a_frag(sl16, 128, e0, 64, lane);
  v16h a3 = load_a_frag(sl16, 128, e0, 96, lane);

  const int n  = lane & 15;
  const int rb = (lane & 16) ? 8 : 0;

  for (int h = wave; h < 32; h += 4) {  // each wave: 8 heads
    float acc[8];
#pragma unroll
    for (int r = 0; r < 8; ++r) acc[r] = 0.0f;
#pragma unroll
    for (int t = 0; t < 2; ++t) {  // head spans two 16-col tiles
      const int n0 = h * 32 + t * 16;
      v8f c = {};
      c = __builtin_amdgcn_wmma_f32_16x16x32_f16(
          false, a0, false, load_b_frag(Wk16, 1024, 0, n0, lane), (short)0, c,
          false, false);
      c = __builtin_amdgcn_wmma_f32_16x16x32_f16(
          false, a1, false, load_b_frag(Wk16, 1024, 32, n0, lane), (short)0, c,
          false, false);
      c = __builtin_amdgcn_wmma_f32_16x16x32_f16(
          false, a2, false, load_b_frag(Wk16, 1024, 64, n0, lane), (short)0, c,
          false, false);
      c = __builtin_amdgcn_wmma_f32_16x16x32_f16(
          false, a3, false, load_b_frag(Wk16, 1024, 96, n0, lane), (short)0, c,
          false, false);
#pragma unroll
      for (int r = 0; r < 8; ++r) {
        const int row = rb + r;
        acc[r] += c[r] * Qn[(size_t)sc[row] * 1024 + n0 + n];
      }
    }
#pragma unroll
    for (int r = 0; r < 8; ++r) {
      float v = wsum16(acc[r]);
      if (n == 0) {
        const int row = rb + r;
        const float w = 5.0f * v;  // int(isqrt(32)) == 5 per reference
        Wsc[(size_t)(e0 + row) * 32 + h] = w;
        atomicMax(&nmax[sc[row] * 32 + h], fkey(w));
      }
    }
  }
}

// ---------------------------------------------------------------------------
__global__ void init_nodes(unsigned* nmax, float* den, float* envn) {
  int i = blockIdx.x * blockDim.x + threadIdx.x;
  if (i < NNODES * 32) { nmax[i] = NEG_INF_KEY; den[i] = 0.0f; }
  if (i < NNODES * 128) envn[i] = 0.0f;
}

__global__ void exp_den(const int* __restrict__ ecen,
                        const unsigned* __restrict__ nmax,
                        float* __restrict__ Wsc, float* __restrict__ den) {
  int i = blockIdx.x * blockDim.x + threadIdx.x;
  if (i >= NEDGES * 32) return;
  int e = i >> 5, m = i & 31;
  int cnode = ecen[e];
  float ex = __expf(Wsc[i] - funkey(nmax[cnode * 32 + m]));
  Wsc[i] = ex;
  atomicAdd(&den[cnode * 32 + m], ex);
}

__global__ void scatter_env(const int* __restrict__ ecen,
                            const float* __restrict__ Wsc,
                            const float* __restrict__ den,
                            const float* __restrict__ g32,
                            const float* __restrict__ equiv,
                            float* __restrict__ envn) {
  int i = blockIdx.x * blockDim.x + threadIdx.x;
  if (i >= NEDGES * 32) return;
  int e = i >> 5, m = i & 31;
  int cnode = ecen[e];
  float attn = Wsc[i] / den[cnode * 32 + m];
  const float* q = equiv + (size_t)e * 128 + m * 4;
  float g0 = g32[(size_t)e * 64 + m] * attn;
  float g1 = g32[(size_t)e * 64 + 32 + m] * attn;
  float* dst = envn + (size_t)cnode * 128 + m * 4;
  atomicAdd(dst + 0, q[0] * g0);
  atomicAdd(dst + 1, q[1] * g1);
  atomicAdd(dst + 2, q[2] * g1);
  atomicAdd(dst + 3, q[3] * g1);
}

__global__ void node_norm(float* envn) {
  const int node = blockIdx.x;
  const int m    = threadIdx.x;  // 0..31 (one wave per node)
  float* p = envn + (size_t)node * 128 + m * 4;
  float x0 = p[0], x1 = p[1], x2 = p[2], x3 = p[3];
  float ss = wsum32(x0 * x0);
  float sv = wsum32(x1 * x1 + x2 * x2 + x3 * x3);
  float rs = rsqrtf(ss * (1.0f / 32.0f) + 1e-6f);
  float rv = rsqrtf(sv * (1.0f / 96.0f) + 1e-6f);
  p[0] = x0 * rs; p[1] = x1 * rv; p[2] = x2 * rv; p[3] = x3 * rv;
}

// ---------------------------------------------------------------------------
// Per-edge tensor products, RMS norms, Ws/Wv einsums and equivariant output.
// One wave per edge (lane = m / lane = k), 4 edges per block.
// ---------------------------------------------------------------------------
__global__ void tensor_product(const float* __restrict__ equiv,
                               const float* __restrict__ envn,
                               const int* __restrict__ ecen,
                               const float* __restrict__ Ws,
                               const float* __restrict__ Wv,
                               const float* __restrict__ eg32,
                               _Float16* __restrict__ scalars16,
                               float* __restrict__ out_equiv) {
  __shared__ float sWs[2048];
  __shared__ float sWv[2048];
  __shared__ float so[4][8][32];  // [wave][o0,o1,o2xyz,o3xyz][m]
  const int wave = threadIdx.x >> 5;
  const int lane = threadIdx.x & 31;
  for (int i = threadIdx.x; i < 2048; i += blockDim.x) {
    sWs[i] = Ws[i];
    sWv[i] = Wv[i];
  }
  const int e     = blockIdx.x * 4 + wave;
  const int cnode = ecen[e];
  const float* x1 = equiv + (size_t)e * 128 + lane * 4;
  const float* x2 = envn + (size_t)cnode * 128 + lane * 4;
  float x1s = x1[0], x1v0 = x1[1], x1v1 = x1[2], x1v2 = x1[3];
  float x2s = x2[0], x2v0 = x2[1], x2v1 = x2[2], x2v2 = x2[3];
  float o0  = x1s * x2s;
  float o1  = (x1v0 * x2v0 + x1v1 * x2v1 + x1v2 * x2v2) * 0.5773502691896258f;
  float o20 = x1s * x2v0, o21 = x1s * x2v1, o22 = x1s * x2v2;
  float o30 = x1v0 * x2s, o31 = x1v1 * x2s, o32 = x1v2 * x2s;
  float r0 = rsqrtf(wsum32(o0 * o0) * (1.0f / 32.0f) + 1e-6f);
  float r1 = rsqrtf(wsum32(o1 * o1) * (1.0f / 32.0f) + 1e-6f);
  float r2 = rsqrtf(wsum32(o20 * o20 + o21 * o21 + o22 * o22) * (1.0f / 96.0f) + 1e-6f);
  float r3 = rsqrtf(wsum32(o30 * o30 + o31 * o31 + o32 * o32) * (1.0f / 96.0f) + 1e-6f);
  o0 *= r0; o1 *= r1;
  o20 *= r2; o21 *= r2; o22 *= r2;
  o30 *= r3; o31 *= r3; o32 *= r3;
  scalars16[(size_t)e * 64 + 2 * lane]     = (_Float16)o0;
  scalars16[(size_t)e * 64 + 2 * lane + 1] = (_Float16)o1;
  so[wave][0][lane] = o0;  so[wave][1][lane] = o1;
  so[wave][2][lane] = o20; so[wave][3][lane] = o21; so[wave][4][lane] = o22;
  so[wave][5][lane] = o30; so[wave][6][lane] = o31; so[wave][7][lane] = o32;
  __syncthreads();
  const int k = lane;
  float s = 0.f, v0 = 0.f, v1 = 0.f, v2 = 0.f;
#pragma unroll 4
  for (int m = 0; m < 32; ++m) {
    float w0 = sWs[m * 32 + k], w1 = sWs[1024 + m * 32 + k];
    s += so[wave][0][m] * w0 + so[wave][1][m] * w1;
    float u0 = sWv[m * 32 + k], u1 = sWv[1024 + m * 32 + k];
    v0 += so[wave][2][m] * u0 + so[wave][5][m] * u1;
    v1 += so[wave][3][m] * u0 + so[wave][6][m] * u1;
    v2 += so[wave][4][m] * u0 + so[wave][7][m] * u1;
  }
  float eg0 = eg32[(size_t)e * 64 + k];
  float egv = eg32[(size_t)e * 64 + 32 + k];
  float* o = out_equiv + (size_t)e * 128 + k * 4;
  o[0] = eg0 * s; o[1] = egv * v0; o[2] = egv * v1; o[3] = egv * v2;
}

// ---------------------------------------------------------------------------
__global__ void fill_upd(const float* __restrict__ sl,
                         const float* __restrict__ ucoef,
                         float* __restrict__ upd) {
  size_t i = blockIdx.x * (size_t)blockDim.x + threadIdx.x;
  if (i >= (size_t)NEDGES * 128) return;
  float u = ucoef[0];
  upd[i] = rsqrtf(u * u + 1.0f) * sl[i];  // c_old * scalar_latent
}

__global__ void layernorm_rows(float* __restrict__ upd,
                               const float* __restrict__ g,
                               const float* __restrict__ b) {
  const int wave = threadIdx.x >> 5, lane = threadIdx.x & 31;
  const int row  = blockIdx.x * 8 + wave;
  float* p = upd + (size_t)row * 128;
  float x0 = p[lane], x1 = p[32 + lane], x2 = p[64 + lane], x3 = p[96 + lane];
  float mu = wsum32(x0 + x1 + x2 + x3) * (1.0f / 128.0f);
  x0 -= mu; x1 -= mu; x2 -= mu; x3 -= mu;
  float var = wsum32(x0 * x0 + x1 * x1 + x2 * x2 + x3 * x3) * (1.0f / 128.0f);
  float rs = rsqrtf(var + 1e-5f);
  p[lane]      = x0 * rs * g[lane]      + b[lane];
  p[32 + lane] = x1 * rs * g[32 + lane] + b[32 + lane];
  p[64 + lane] = x2 * rs * g[64 + lane] + b[64 + lane];
  p[96 + lane] = x3 * rs * g[96 + lane] + b[96 + lane];
}

__global__ void f32_to_f16(const float* __restrict__ s,
                           _Float16* __restrict__ d, size_t n) {
  size_t i = blockIdx.x * (size_t)blockDim.x + threadIdx.x;
  if (i < n) d[i] = (_Float16)s[i];
}

// ---------------------------------------------------------------------------
extern "C" void kernel_launch(void* const* d_in, const int* in_sizes, int n_in,
                              void* d_out, int out_size, void* d_ws,
                              size_t ws_size, hipStream_t stream) {
  (void)in_sizes; (void)n_in; (void)out_size; (void)ws_size;

  const float* node_inv = (const float*)d_in[0];
  const float* sl       = (const float*)d_in[1];
  const float* equiv    = (const float*)d_in[2];
  const float* cond     = (const float*)d_in[3];
  const float* ucoef    = (const float*)d_in[4];
  const int*   ecen     = (const int*)d_in[5];
  const int*   activ    = (const int*)d_in[6];
  const float* envW0 = (const float*)d_in[7];
  const float* envW1 = (const float*)d_in[8];
  const float* envW2 = (const float*)d_in[9];
  const float* envB0 = (const float*)d_in[10];
  const float* envB1 = (const float*)d_in[11];
  const float* envB2 = (const float*)d_in[12];
  const float* Wq    = (const float*)d_in[13];
  const float* Wk    = (const float*)d_in[14];
  const float* sclW0 = (const float*)d_in[15];
  const float* sclW1 = (const float*)d_in[16];
  const float* sclW2 = (const float*)d_in[17];
  const float* sclB0 = (const float*)d_in[18];
  const float* sclB1 = (const float*)d_in[19];
  const float* sclB2 = (const float*)d_in[20];
  const float* lng   = (const float*)d_in[21];
  const float* lnb   = (const float*)d_in[22];
  const float* eqW0  = (const float*)d_in[23];
  const float* eqW1  = (const float*)d_in[24];
  const float* eqW2  = (const float*)d_in[25];
  const float* eqB0  = (const float*)d_in[26];
  const float* eqB1  = (const float*)d_in[27];
  const float* eqB2  = (const float*)d_in[28];
  const float* WsM   = (const float*)d_in[29];
  const float* WvM   = (const float*)d_in[30];

  char* ws = (char*)d_ws;
  size_t off = 0;
  auto take = [&](size_t bytes) -> char* {
    char* p = ws + off;
    off += (bytes + 255) & ~(size_t)255;
    return p;
  };

  _Float16* w16pool  = (_Float16*)take((size_t)315392 * 2);
  _Float16* envW16_0 = w16pool + 0;       // 160x128
  _Float16* envW16_1 = w16pool + 20480;   // 128x128
  _Float16* envW16_2 = w16pool + 36864;   // 128x64
  _Float16* Wq16     = w16pool + 45056;   // 64x1024
  _Float16* Wk16     = w16pool + 110592;  // 128x1024
  _Float16* sclW16_0 = w16pool + 241664;  // 96x128
  _Float16* sclW16_1 = w16pool + 253952;  // 128x128
  _Float16* sclW16_2 = w16pool + 270336;  // 128x128
  _Float16* eqW16_0  = w16pool + 286720;  // 32x128
  _Float16* eqW16_1  = w16pool + 290816;  // 128x128
  _Float16* eqW16_2  = w16pool + 307200;  // 128x64

  _Float16* sl16   = (_Float16*)take((size_t)NEDGES * 128 * 2);
  _Float16* cond16 = (_Float16*)take((size_t)NEDGES * 32 * 2);
  _Float16* ni16   = (_Float16*)take((size_t)NNODES * 64 * 2);
  float*    Qn32   = (float*)take((size_t)NNODES * 1024 * 4);
  float*    Wsc    = (float*)take((size_t)NEDGES * 32 * 4);
  unsigned* nmax   = (unsigned*)take((size_t)NNODES * 32 * 4);
  float*    den    = (float*)take((size_t)NNODES * 32 * 4);
  float*    envn   = (float*)take((size_t)NNODES * 128 * 4);
  _Float16* h16a   = (_Float16*)take((size_t)NEDGES * 128 * 2);
  _Float16* h16b   = (_Float16*)take((size_t)NEDGES * 128 * 2);
  float*    g32    = (float*)take((size_t)NEDGES * 64 * 4);  // env gate, reused as eq gate
  _Float16* sca16  = (_Float16*)take((size_t)NEDGES * 64 * 2);

  float* out_scalar = (float*)d_out;                    // E x 128
  float* out_equiv  = out_scalar + (size_t)NEDGES * 128;  // E x 32 x 4

  auto cvt = [&](const float* s, _Float16* d, size_t n) {
    f32_to_f16<<<dim3((unsigned)((n + 255) / 256)), dim3(256), 0, stream>>>(s, d, n);
  };

  // 1) precision conversion of activations + weights
  cvt(sl, sl16, (size_t)NEDGES * 128);
  cvt(cond, cond16, (size_t)NEDGES * 32);
  cvt(node_inv, ni16, (size_t)NNODES * 64);
  cvt(envW0, envW16_0, 160 * 128);
  cvt(envW1, envW16_1, 128 * 128);
  cvt(envW2, envW16_2, 128 * 64);
  cvt(Wq, Wq16, 64 * 1024);
  cvt(Wk, Wk16, 128 * 1024);
  cvt(sclW0, sclW16_0, 96 * 128);
  cvt(sclW1, sclW16_1, 128 * 128);
  cvt(sclW2, sclW16_2, 128 * 128);
  cvt(eqW0, eqW16_0, 32 * 128);
  cvt(eqW1, eqW16_1, 128 * 128);
  cvt(eqW2, eqW16_2, 128 * 64);

  init_nodes<<<dim3((NNODES * 128 + 255) / 256), dim3(256), 0, stream>>>(nmax, den, envn);
  fill_upd<<<dim3((unsigned)(((size_t)NEDGES * 128 + 255) / 256)), dim3(256), 0, stream>>>(
      sl, ucoef, out_scalar);

  // 2) Qn = node_invariants @ Wq  (per-node instead of per-edge: 32x fewer FLOPs)
  gemm_wmma<0, 1><<<dim3(NNODES / 16, 1024 / 64), dim3(128), 0, stream>>>(
      ni16, 64, nullptr, 0, Wq16, nullptr, 1024, nullptr, Qn32, nullptr, nullptr);

  // 3) env MLP: [sl|cond] -> 128 -> 128 -> 64 (last layer linear, f32)
  gemm_wmma<1, 0><<<dim3(NEDGES / 16, 2), dim3(128), 0, stream>>>(
      sl16, 128, cond16, 32, envW16_0, envB0, 128, h16a, nullptr, nullptr, nullptr);
  gemm_wmma<1, 0><<<dim3(NEDGES / 16, 2), dim3(128), 0, stream>>>(
      h16a, 128, nullptr, 0, envW16_1, envB1, 128, h16b, nullptr, nullptr, nullptr);
  gemm_wmma<0, 1><<<dim3(NEDGES / 16, 1), dim3(128), 0, stream>>>(
      h16b, 128, nullptr, 0, envW16_2, envB2, 64, nullptr, g32, nullptr, nullptr);

  // 4) fused QK scores (K never materialized) + segment max
  score_wmma<<<dim3(NEDGES / 16), dim3(128), 0, stream>>>(sl16, Wk16, Qn32, ecen, Wsc, nmax);

  // 5) softmax numerator/denominator
  exp_den<<<dim3((NEDGES * 32 + 255) / 256), dim3(256), 0, stream>>>(ecen, nmax, Wsc, den);

  // 6) attention-weighted scatter of env_edges into nodes
  scatter_env<<<dim3((NEDGES * 32 + 255) / 256), dim3(256), 0, stream>>>(
      ecen, Wsc, den, g32, equiv, envn);

  // 7) node RMS norms (scalar over M, vector over M*3)
  node_norm<<<dim3(NNODES), dim3(32), 0, stream>>>(envn);

  // 8) eq MLP (gates) -> reuse g32 as eg
  gemm_wmma<1, 0><<<dim3(NEDGES / 16, 2), dim3(128), 0, stream>>>(
      cond16, 32, nullptr, 0, eqW16_0, eqB0, 128, h16a, nullptr, nullptr, nullptr);
  gemm_wmma<1, 0><<<dim3(NEDGES / 16, 2), dim3(128), 0, stream>>>(
      h16a, 128, nullptr, 0, eqW16_1, eqB1, 128, h16b, nullptr, nullptr, nullptr);
  gemm_wmma<0, 1><<<dim3(NEDGES / 16, 1), dim3(128), 0, stream>>>(
      h16b, 128, nullptr, 0, eqW16_2, eqB2, 64, nullptr, g32, nullptr, nullptr);

  // 9) tensor products + Ws/Wv einsums -> equivariant output + scalars16
  tensor_product<<<dim3(NEDGES / 4), dim3(128), 0, stream>>>(
      equiv, envn, ecen, WsM, WvM, g32, sca16, out_equiv);

  // 10) scl MLP; last layer scatter-adds c_new*out into upd rows active[e]
  gemm_wmma<1, 0><<<dim3(NEDGES / 16, 2), dim3(128), 0, stream>>>(
      sca16, 64, cond16, 32, sclW16_0, sclB0, 128, h16a, nullptr, nullptr, nullptr);
  gemm_wmma<1, 0><<<dim3(NEDGES / 16, 2), dim3(128), 0, stream>>>(
      h16a, 128, nullptr, 0, sclW16_1, sclB1, 128, h16b, nullptr, nullptr, nullptr);
  gemm_wmma<0, 2><<<dim3(NEDGES / 16, 2), dim3(128), 0, stream>>>(
      h16b, 128, nullptr, 0, sclW16_2, sclB2, 128, nullptr, out_scalar, activ, ucoef);

  // 11) LayerNorm in place on the scalar output
  layernorm_rows<<<dim3(NEDGES / 8), dim3(256), 0, stream>>>(out_scalar, lng, lnb);
}
